// CrossAttention_88622355186401
// MI455X (gfx1250) — compile-verified, matrix operand order checked by reading, not run on hardware
//
#include <hip/hip_runtime.h>

typedef _Float16 v16h __attribute__((ext_vector_type(16)));
typedef _Float16 h2   __attribute__((ext_vector_type(2)));
typedef float    v8f  __attribute__((ext_vector_type(8)));
typedef unsigned int u32x4 __attribute__((ext_vector_type(4)));
typedef int          i32x8 __attribute__((ext_vector_type(8)));
typedef int          i32x4 __attribute__((ext_vector_type(4)));

#define DIM    128
#define HEADS  4
#define HD     32
#define NW     256      // windows total (B * 8 * 8)
#define LQ     128      // q tokens per window
#define LK     512      // k tokens per window
#define SCALE  0.17677669529663687f  // 32^-0.5

// ---------------------------------------------------------------------------
// Fragment loaders for v_wmma_f32_16x16x32_f16 (gfx1250 layout):
// A (16x32 f16): lane m (0..15) holds row m, K = {0..7, 16..23};
//                lane m+16 holds row m, K = {8..15, 24..31}.
// B (32x16 f16): lane n holds column n with the same K striping -> a
//                row-major K-vector per lane, i.e. B = (k rows)^T for free.
// ---------------------------------------------------------------------------
__device__ __forceinline__ v16h frag_h(const _Float16* base, int stride, int lane) {
    int m = lane & 15, s = (lane >> 4) & 1;
    const _Float16* r = base + m * stride;
    v16h o;
#pragma unroll
    for (int t = 0; t < 8; ++t) o[t] = r[s * 8 + t];
#pragma unroll
    for (int t = 0; t < 8; ++t) o[8 + t] = r[16 + s * 8 + t];
    return o;
}

__device__ __forceinline__ v16h frag_f(const float* base, int stride, int lane) {
    int m = lane & 15, s = (lane >> 4) & 1;
    const float* r = base + m * stride;
    v16h o;
#pragma unroll
    for (int t = 0; t < 8; ++t) o[t] = (_Float16)r[s * 8 + t];
#pragma unroll
    for (int t = 0; t < 8; ++t) o[8 + t] = (_Float16)r[16 + s * 8 + t];
    return o;
}

__device__ __forceinline__ v8f wmma_f16(v16h a, v16h b, v8f c) {
    return __builtin_amdgcn_wmma_f32_16x16x32_f16(false, a, false, b, (short)0, c, false, false);
}

// packed-f16 32-wide dot product (v_pk_fma_f16 chain)
__device__ __forceinline__ float dot32h(const _Float16* a, const _Float16* b) {
    h2 acc = { (_Float16)0.f, (_Float16)0.f };
#pragma unroll
    for (int t = 0; t < 16; ++t) {
        h2 av = *(const h2*)(a + 2 * t);
        h2 bv = *(const h2*)(b + 2 * t);
        acc = av * bv + acc;
    }
    return (float)acc[0] + (float)acc[1];
}

// ---------------------------------------------------------------------------
// Tensor Data Mover: 1-D contiguous global->LDS DMA (f16 elements).
// D# per CDNA5 ISA ch.8: group0 {count, lds_addr, global_addr, type=2},
// group1 {data_size=2B, tensor_dim0=n, tensor_dim1=1, tile_dim0=n, tile_dim1=1,
//         stride0=n}. Groups 2/3 zero (<=2D tensor).
// This toolchain exposes the 6-arg builtin (g0 u32x4, g1 i32x8, g2 i32x4,
// g3 i32x4, ext i32x8, cpol i32).
// ---------------------------------------------------------------------------
#if __has_builtin(__builtin_amdgcn_tensor_load_to_lds) && __has_builtin(__builtin_amdgcn_s_wait_tensorcnt)
#define USE_TDM 1
__device__ __forceinline__ void tdm_load_1d(const void* gsrc, unsigned lds_off, unsigned nelem) {
    unsigned long long ga = (unsigned long long)(size_t)gsrc;
    u32x4 g0;
    g0[0] = 1u;                                             // count=1, user mode
    g0[1] = lds_off;                                        // lds_addr [63:32]
    g0[2] = (unsigned)(ga & 0xFFFFFFFFu);                   // global_addr lo
    g0[3] = (unsigned)((ga >> 32) & 0x1FFFFFFu) | (2u << 30); // addr hi + type=2
    i32x8 g1;
    g1[0] = (int)(1u << 16);                                // data_size=1 (2 bytes)
    g1[1] = (int)((nelem & 0xFFFFu) << 16);                 // tensor_dim0 lo16 @ [63:48]
    g1[2] = (int)((nelem >> 16) | (1u << 16));              // dim0 hi16, tensor_dim1=1
    g1[3] = (int)((nelem & 0xFFFFu) << 16);                 // tile_dim0 @ [127:112]
    g1[4] = 1;                                              // tile_dim1=1, tile_dim2=0
    g1[5] = (int)nelem;                                     // tensor_dim0_stride lo32
    g1[6] = 0;                                              // stride0 hi, stride1 lo
    g1[7] = 0;
    i32x4 z4 = { 0, 0, 0, 0 };
    i32x8 z8 = { 0, 0, 0, 0, 0, 0, 0, 0 };
    __builtin_amdgcn_tensor_load_to_lds(g0, g1, z4, z4, z8, 0);
}
#else
#define USE_TDM 0
#endif

// ---------------------------------------------------------------------------
// Kernel 1: q = (x @ Wq^T + bq) * SCALE -> f16, window-partitioned [w][h][i][c]
// 256 threads = 8 independent wave-tiles of 16 tokens x 16 dims, K=128.
// ---------------------------------------------------------------------------
__global__ void __launch_bounds__(256)
k_proj_q(const float* __restrict__ x, const float* __restrict__ Wq,
         const float* __restrict__ bq, _Float16* __restrict__ qf) {
    int lane = threadIdx.x & 31;
    int gt = blockIdx.x * 8 + (threadIdx.x >> 5);
    int tt = gt >> 3;   // token tile (32768/16)
    int dt = gt & 7;    // dim tile (128/16)
    v8f acc = {};
#pragma unroll
    for (int ks = 0; ks < 4; ++ks) {
        v16h a = frag_f(x  + (tt * 16) * DIM + ks * 32, DIM, lane);
        v16h b = frag_f(Wq + (dt * 16) * DIM + ks * 32, DIM, lane);
        acc = wmma_f16(a, b, acc);
    }
    int n = lane & 15, hi = lane >> 4;
    int d = dt * 16 + n;
    float bias = bq[d];
    int h = d >> 5, c = d & 31;
#pragma unroll
    for (int r = 0; r < 8; ++r) {
        int t  = tt * 16 + r + 8 * hi;
        int b0 = t >> 13;
        int t2 = t & 8191;
        int y  = t2 >> 7;
        int t3 = t2 & 127;
        int xc = t3 >> 1;
        int m  = t3 & 1;
        int w  = b0 * 64 + (y >> 3) * 8 + (xc >> 3);
        int i  = (((y & 7) * 8 + (xc & 7)) << 1) + m;
        float v = (acc[r] + bias) * SCALE;
        qf[((w * HEADS + h) * LQ + i) * HD + c] = (_Float16)v;
    }
}

// ---------------------------------------------------------------------------
// Kernel 2: kv = context @ Wkv^T + bkv -> k,v f16 window-partitioned [w][h][j][c]
// ---------------------------------------------------------------------------
__global__ void __launch_bounds__(256)
k_proj_kv(const float* __restrict__ ctx, const float* __restrict__ Wkv,
          const float* __restrict__ bkv,
          _Float16* __restrict__ kf, _Float16* __restrict__ vf) {
    int lane = threadIdx.x & 31;
    int gt = blockIdx.x * 8 + (threadIdx.x >> 5);
    int tt = gt >> 4;   // token tile (131072/16)
    int dt = gt & 15;   // dim tile (256/16)
    v8f acc = {};
#pragma unroll
    for (int ks = 0; ks < 4; ++ks) {
        v16h a = frag_f(ctx + (tt * 16) * DIM + ks * 32, DIM, lane);
        v16h b = frag_f(Wkv + (dt * 16) * DIM + ks * 32, DIM, lane);
        acc = wmma_f16(a, b, acc);
    }
    int n = lane & 15, hi = lane >> 4;
    int d = dt * 16 + n;
    float bias = bkv[d];
#pragma unroll
    for (int r = 0; r < 8; ++r) {
        int t  = tt * 16 + r + 8 * hi;
        int b0 = t >> 15;
        int t2 = t & 32767;
        int y  = t2 >> 8;
        int t3 = t2 & 255;
        int xc = t3 >> 1;
        int nn = t3 & 1;
        int w  = b0 * 64 + (y >> 4) * 8 + (xc >> 4);
        int j  = (((y & 15) * 16 + (xc & 15)) << 1) + nn;
        float v = acc[r] + bias;
        if (d < DIM) {
            kf[((w * HEADS + (d >> 5)) * LK + j) * HD + (d & 31)] = (_Float16)v;
        } else {
            int d2 = d - DIM;
            vf[((w * HEADS + (d2 >> 5)) * LK + j) * HD + (d2 & 31)] = (_Float16)v;
        }
    }
}

// ---------------------------------------------------------------------------
// Kernel 3: gather cRPE tables (dedup over M/N repeats).
// idx(p=a*8+c, jk=b*16+d) = 30*(2a-b+15) + (2c-d+15); row = [HEADS][96] f32.
// q_rpe pre-scaled by SCALE. v_rpe stored transposed [h][p][c][jk].
// ---------------------------------------------------------------------------
__global__ void k_rpe(const float* __restrict__ table,
                      _Float16* __restrict__ qrpe, _Float16* __restrict__ krpe,
                      _Float16* __restrict__ vrpeT) {
    int t  = blockIdx.x * blockDim.x + threadIdx.x;   // 4*64*256 = 65536
    int jk = t & 255;
    int p  = (t >> 8) & 63;
    int h  = t >> 14;
    int a  = p >> 3, cq = p & 7;
    int bb = jk >> 4, dd = jk & 15;
    int row = 30 * (2 * a - bb + 15) + (2 * cq - dd + 15);
    const float* src = table + row * (3 * DIM) + h * 96;
    int base = ((h * 64 + p) * 256 + jk) * HD;
#pragma unroll
    for (int c = 0; c < HD; ++c) {
        qrpe[base + c] = (_Float16)(src[c] * SCALE);
        krpe[base + c] = (_Float16)src[32 + c];
        vrpeT[((h * 64 + p) * HD + c) * 256 + jk] = (_Float16)src[64 + c];
    }
}

// ---------------------------------------------------------------------------
// Kernel 4: windowed flash attention with cRPE.
// block = (window, head), 256 threads = 8 waves, each wave owns 16 q rows.
// K and Q tiles are contiguous in the workspace -> staged by TDM DMA (wave 0),
// V is staged transposed by all threads in parallel with the DMA.
// ---------------------------------------------------------------------------
__global__ void __launch_bounds__(256)
k_attn(const _Float16* __restrict__ qf, const _Float16* __restrict__ kf,
       const _Float16* __restrict__ vf,
       const _Float16* __restrict__ qrpe, const _Float16* __restrict__ krpe,
       const _Float16* __restrict__ vrpeT, _Float16* __restrict__ ao) {
    __shared__ __align__(16) _Float16 k_lds[LK * HD];     // 32 KB  [j][c]
    __shared__ __align__(16) _Float16 vT_lds[HD * LK];    // 32 KB  [c][j]
    __shared__ __align__(16) _Float16 q_lds[LQ * HD];     // 8 KB   [i][c]
    __shared__ __align__(16) _Float16 p_lds[8 * 16 * 32]; // 8 KB   per-wave P tile

    int w = blockIdx.x >> 2, h = blockIdx.x & 3;
    int tid = threadIdx.x, lane = tid & 31, wv = tid >> 5;

    const _Float16* kg = kf + ((w * HEADS + h) * LK) * HD;
    const _Float16* vg = vf + ((w * HEADS + h) * LK) * HD;
    const _Float16* qg = qf + ((w * HEADS + h) * LQ) * HD;

#if USE_TDM
    if (wv == 0) {
        tdm_load_1d(kg, (unsigned)(size_t)(void*)k_lds, LK * HD);
        tdm_load_1d(qg, (unsigned)(size_t)(void*)q_lds, LQ * HD);
        __builtin_amdgcn_s_wait_tensorcnt(0);
    }
#else
    for (int r = tid; r < LK; r += 256) {
        const uint4* s = (const uint4*)(kg + r * HD);
        uint4* d = (uint4*)(k_lds + r * HD);
        d[0] = s[0]; d[1] = s[1]; d[2] = s[2]; d[3] = s[3];
    }
    for (int r2 = tid; r2 < LQ * 2; r2 += 256) {
        int rr = r2 >> 1, hh = r2 & 1;
        const uint4* s = (const uint4*)(qg + rr * HD + hh * 16);
        uint4* d = (uint4*)(q_lds + rr * HD + hh * 16);
        d[0] = s[0]; d[1] = s[1];
    }
#endif
    // stage V transposed (feeds P*V B fragments with contiguous loads)
    for (int r = tid; r < LK; r += 256) {
        const _Float16* s = vg + r * HD;
#pragma unroll
        for (int c = 0; c < HD; ++c) vT_lds[c * LK + r] = s[c];
    }
    __syncthreads();

    int i0 = wv * 16;
    int n = lane & 15, hi = lane >> 4;
    v16h a_q = frag_h(q_lds + i0 * HD, HD, lane);

    float mrow[8], lrow[8];
    v8f o0 = {}, o1 = {};
#pragma unroll
    for (int r = 0; r < 8; ++r) { mrow[r] = -1e30f; lrow[r] = 0.f; }

    _Float16* prow_base = p_lds + wv * 16 * 32;

    for (int jb = 0; jb < 16; ++jb) {
        int jcol0 = jb * 32;
        // ---- S = q @ k^T (two 16x16 n-tiles, K=32) ----
        v16h bk0 = frag_h(k_lds + jcol0 * HD, HD, lane);
        v16h bk1 = frag_h(k_lds + (jcol0 + 16) * HD, HD, lane);
        v8f zero = {};
        v8f s0 = wmma_f16(a_q, bk0, zero);
        v8f s1 = wmma_f16(a_q, bk1, zero);

        int j0 = jcol0 + n, j1 = jcol0 + 16 + n;

        // ---- qr = q . k_rpe : jk = j>>1 shared by lane pairs (n, n^1).
        //      Even lane computes the s0 column, odd lane the s1 column,
        //      then exchange with a 1-lane shuffle. 8 dots instead of 16. ----
        int jkm = (jcol0 >> 1) + ((n & 1) ? 8 : 0) + (n >> 1);
        const _Float16* krm_base = krpe + (h * 64) * 256 * HD;
#pragma unroll
        for (int r = 0; r < 8; ++r) {
            int i = i0 + r + 8 * hi;
            int p = i >> 1;
            float qv = dot32h(q_lds + i * HD, krm_base + (p * 256 + jkm) * HD);
            float qo = __shfl_xor(qv, 1, 32);
            s0[r] += (n & 1) ? qo : qv;
            s1[r] += (n & 1) ? qv : qo;
        }

        // ---- kr = k . q_rpe : p = i>>1 shared by row pairs (r, r+1).
        //      8 dots instead of 16. ----
        const _Float16* krow0 = k_lds + j0 * HD;
        const _Float16* krow1 = k_lds + j1 * HD;
#pragma unroll
        for (int re = 0; re < 4; ++re) {
            int p = (i0 + 2 * re + 8 * hi) >> 1;
            int rpb = (h * 64 + p) * 256;
            float kr0 = dot32h(krow0, qrpe + (rpb + (j0 >> 1)) * HD);
            float kr1 = dot32h(krow1, qrpe + (rpb + (j1 >> 1)) * HD);
            s0[2 * re] += kr0; s0[2 * re + 1] += kr0;
            s1[2 * re] += kr1; s1[2 * re + 1] += kr1;
        }

        // ---- online softmax over the 32-column block ----
#pragma unroll
        for (int r = 0; r < 8; ++r) {
            float mx = fmaxf(s0[r], s1[r]);
#pragma unroll
            for (int msk = 8; msk > 0; msk >>= 1)
                mx = fmaxf(mx, __shfl_xor(mx, msk, 32));
            float mnew = fmaxf(mrow[r], mx);
            float sc = __expf(mrow[r] - mnew);
            float p0 = __expf(s0[r] - mnew);
            float p1 = __expf(s1[r] - mnew);
            float rs = p0 + p1;
#pragma unroll
            for (int msk = 8; msk > 0; msk >>= 1)
                rs += __shfl_xor(rs, msk, 32);
            lrow[r] = lrow[r] * sc + rs;
            mrow[r] = mnew;
            o0[r] *= sc; o1[r] *= sc;
            prow_base[(r + 8 * hi) * 32 + n]      = (_Float16)p0;
            prow_base[(r + 8 * hi) * 32 + 16 + n] = (_Float16)p1;
        }

        // ---- O += P @ V (A-layout P via LDS round-trip, transposed V frags) ----
        v16h a_p = frag_h(prow_base, 32, lane);
        v16h bv0 = frag_h(vT_lds + jcol0, LK, lane);
        v16h bv1 = frag_h(vT_lds + 16 * LK + jcol0, LK, lane);
        o0 = wmma_f16(a_p, bv0, o0);
        o1 = wmma_f16(a_p, bv1, o1);

        // ---- O += P @ v_rpe : rpe is N-repeat invariant (pair-sum P) and
        //      M-repeat invariant (hoist vr rows over row pairs). ----
#pragma unroll
        for (int re = 0; re < 4; ++re) {
            int p = (i0 + 2 * re + 8 * hi) >> 1;
            const _Float16* vr0 = vrpeT + ((h * 64 + p) * HD + n) * 256 + jb * 16;
            const _Float16* vr1 = vr0 + 16 * 256;
#pragma unroll
            for (int rr = 0; rr < 2; ++rr) {
                int r = 2 * re + rr;
                const _Float16* pr = prow_base + (r + 8 * hi) * 32;
                float a0 = 0.f, a1 = 0.f;
#pragma unroll
                for (int kk = 0; kk < 16; ++kk) {
                    float ps = (float)pr[2 * kk] + (float)pr[2 * kk + 1];
                    a0 += ps * (float)vr0[kk];
                    a1 += ps * (float)vr1[kk];
                }
                o0[r] += a0; o1[r] += a1;
            }
        }
    }

    // ---- normalize and store f16 [w][h][i][c] ----
    _Float16* aob = ao + ((w * HEADS + h) * LQ) * HD;
#pragma unroll
    for (int r = 0; r < 8; ++r) {
        int i = i0 + r + 8 * hi;
        float inv = 1.f / lrow[r];
        aob[i * HD + n]      = (_Float16)(o0[r] * inv);
        aob[i * HD + 16 + n] = (_Float16)(o1[r] * inv);
    }
}

// ---------------------------------------------------------------------------
// Kernel 5: out = attn_out @ Wproj^T + bproj, scatter to (B,H,W,M,DIM) fp32
// ---------------------------------------------------------------------------
__global__ void __launch_bounds__(256)
k_proj_o(const _Float16* __restrict__ ao, const float* __restrict__ Wp,
         const float* __restrict__ bp, float* __restrict__ out) {
    int lane = threadIdx.x & 31;
    int gt = blockIdx.x * 8 + (threadIdx.x >> 5);
    int tt = gt >> 3;
    int dt = gt & 7;
    int m = lane & 15, s = (lane >> 4) & 1;
    int tA = tt * 16 + m;
    int wA = tA >> 7, iA = tA & 127;
    v8f acc = {};
#pragma unroll
    for (int ks = 0; ks < 4; ++ks) {   // K block ks == head (layout [w][h][i][c])
        const _Float16* r = ao + ((wA * HEADS + ks) * LQ + iA) * HD;
        v16h a;
#pragma unroll
        for (int u = 0; u < 8; ++u) a[u] = r[s * 8 + u];
#pragma unroll
        for (int u = 0; u < 8; ++u) a[8 + u] = r[16 + s * 8 + u];
        v16h b = frag_f(Wp + (dt * 16) * DIM + ks * 32, DIM, lane);
        acc = wmma_f16(a, b, acc);
    }
    int n = lane & 15, hi = lane >> 4;
    int d = dt * 16 + n;
    float bias = bp[d];
#pragma unroll
    for (int r = 0; r < 8; ++r) {
        int t2 = tt * 16 + r + 8 * hi;
        int w2 = t2 >> 7, i2 = t2 & 127;
        int b0 = w2 >> 6, wy = (w2 >> 3) & 7, wx = w2 & 7;
        int ih = i2 >> 1, mm = i2 & 1;
        int r8 = ih >> 3, c8 = ih & 7;
        int y = wy * 8 + r8, xx = wx * 8 + c8;
        long oidx = ((((long)b0 * 64 + y) * 64 + xx) * 2 + mm) * DIM + d;
        out[oidx] = acc[r] + bias;
    }
}

// ---------------------------------------------------------------------------
extern "C" void kernel_launch(void* const* d_in, const int* in_sizes, int n_in,
                              void* d_out, int out_size, void* d_ws, size_t ws_size,
                              hipStream_t stream) {
    const float* x   = (const float*)d_in[0];
    const float* ctx = (const float*)d_in[1];
    const float* Wq  = (const float*)d_in[2];
    const float* bq  = (const float*)d_in[3];
    const float* Wkv = (const float*)d_in[4];
    const float* bkv = (const float*)d_in[5];
    const float* Wp  = (const float*)d_in[6];
    const float* bp  = (const float*)d_in[7];
    const float* tab = (const float*)d_in[8];

    char* ws = (char*)d_ws;
    _Float16* qf    = (_Float16*)(ws);                 // 256*4*128*32*2  =  8 MB
    _Float16* kf    = (_Float16*)(ws + 8388608);       // 256*4*512*32*2  = 32 MB
    _Float16* vf    = (_Float16*)(ws + 41943040);      // 32 MB
    _Float16* qrpe  = (_Float16*)(ws + 75497472);      // 4*64*256*32*2   =  4 MB
    _Float16* krpe  = (_Float16*)(ws + 79691776);      // 4 MB
    _Float16* vrpeT = (_Float16*)(ws + 83886080);      // 4 MB
    _Float16* ao    = (_Float16*)(ws + 88080384);      // 8 MB  (total ~96.5 MB)

    k_proj_q <<<2048, 256, 0, stream>>>(x, Wq, bq, qf);
    k_proj_kv<<<16384, 256, 0, stream>>>(ctx, Wkv, bkv, kf, vf);
    k_rpe    <<<256, 256, 0, stream>>>(tab, qrpe, krpe, vrpeT);
    k_attn   <<<1024, 256, 0, stream>>>(qf, kf, vf, qrpe, krpe, vrpeT, ao);
    k_proj_o <<<2048, 256, 0, stream>>>(ao, Wp, bp, (float*)d_out);
}
